// TransformerAutoencoder_8375186227215
// MI455X (gfx1250) — compile-verified, matrix operand order checked by reading, not run on hardware
//
#include <hip/hip_runtime.h>
#include <hip/hip_bf16.h>

typedef __attribute__((ext_vector_type(16))) _Float16 v16h;
typedef __attribute__((ext_vector_type(8)))  _Float16 v8h;
typedef __attribute__((ext_vector_type(8)))  float    v8f;

#define BATCHN 4
#define TSEQ   256     // LSTM input sequence
#define DMODEL 256
#define DTM    512     // transformer model dim
#define FFH    2048
#define SEQ    33      // MAX_SEQ + 1
#define MSTEPS 32
#define NVOCAB 388
#define NHEADS 2
#define HDIM   256

__device__ __forceinline__ float sigmf(float x) { return 1.f / (1.f + expf(-x)); }

#define SHUF16(a, b) __builtin_shufflevector(a, b, 0,1,2,3,4,5,6,7,8,9,10,11,12,13,14,15)

// ---------------------------------------------------------------------------
// Generic WMMA GEMM:  C[M,N] = act(A[M,K](f16) * Wh[Npad,K](f16)^T + bias)
// One 16x16 tile per wave, 4 waves/block. Wh rows padded to ceil(N/16)*16.
// Per-lane operands are two contiguous 16B chunks -> b128 loads, no guards
// in the K loop (row index clamped; garbage rows masked at store).
// ---------------------------------------------------------------------------
__global__ void gemm_wmma(const _Float16* __restrict__ A,
                          const _Float16* __restrict__ Wh,
                          const float* __restrict__ bias,
                          float* __restrict__ C32,
                          _Float16* __restrict__ C16,
                          int M, int N, int K, int relu)
{
  const int lane  = threadIdx.x & 31;
  const int wave  = threadIdx.x >> 5;
  const int ntile = blockIdx.x * 4 + wave;
  const int mtile = blockIdx.y;
  const int nTiles = (N + 15) >> 4;
  if (ntile >= nTiles) return;           // wave-uniform

  const int hi   = lane >> 4;
  const int mrow = mtile * 16 + (lane & 15);
  const int ncol = ntile * 16 + (lane & 15);
  const int mr   = (mrow < M) ? mrow : (M - 1);   // clamp, no branchy loads
  const bool nok = (ncol < N);

  const _Float16* __restrict__ Ap = A  + (size_t)mr   * K + hi * 8;
  const _Float16* __restrict__ Bp = Wh + (size_t)ncol * K + hi * 8;

  v8f acc = {};
  for (int kb = 0; kb < K; kb += 32) {
    const v8h a0 = *(const v8h*)(Ap + kb);
    const v8h a1 = *(const v8h*)(Ap + kb + 16);
    const v8h b0 = *(const v8h*)(Bp + kb);
    const v8h b1 = *(const v8h*)(Bp + kb + 16);
    const v16h av = SHUF16(a0, a1);
    const v16h bv = SHUF16(b0, b1);
    acc = __builtin_amdgcn_wmma_f32_16x16x32_f16(false, av, false, bv,
                                                 (short)0, acc, false, false);
  }
#pragma unroll
  for (int r = 0; r < 8; ++r) {
    const int m = mtile * 16 + r + 8 * hi;
    if (m < M && nok) {
      float v = acc[r];
      if (bias) v += bias[ncol];
      if (relu) v = v > 0.f ? v : 0.f;
      if (C32) C32[(size_t)m * N + ncol] = v;
      if (C16) C16[(size_t)m * N + ncol] = (_Float16)v;
    }
  }
}

// ---------------------------------------------------------------------------
// Persistent LSTM recurrence. grid.x = 2 (dir 0 = fwd, 1 = bwd), 512 threads.
// G = precomputed x@Wih^T + b, rows (b*T + t) x 1024.  h(f16), c(f32) in LDS.
// Each step: z = h @ Whh^T (WMMA, M padded 4->16) + G[t]; gate update.
// A (hidden-state) tile is loaded from LDS once per step into registers.
// ---------------------------------------------------------------------------
__global__ void lstm_rec(const float* __restrict__ Gf,
                         const float* __restrict__ Gb,
                         const _Float16* __restrict__ WhhF,  // (1024,256) f16
                         const _Float16* __restrict__ WhhB,
                         _Float16* __restrict__ x1out,       // (B,T,512) or null
                         _Float16* __restrict__ hs16,        // (B,1024)
                         int hsBase, int T)
{
  __shared__ float zbuf[4][1024];
  __shared__ float cst[4][256];
  __shared__ __align__(16) _Float16 hst[4][256];

  const int dir = blockIdx.x;
  const float*    __restrict__ G   = dir ? Gb : Gf;
  const _Float16* __restrict__ Whh = dir ? WhhB : WhhF;

  const int tid  = threadIdx.x;          // 512 threads = 16 waves
  const int lane = tid & 31;
  const int wave = tid >> 5;
  const int hi   = lane >> 4;
  const int ml   = lane & 15;
  const int mra  = (ml < 4) ? ml : 3;    // clamp batch row

  for (int e = tid; e < 1024; e += 512) {
    cst[e >> 8][e & 255] = 0.f;
    hst[e >> 8][e & 255] = (_Float16)0;
  }
  __syncthreads();

  for (int s = 0; s < T; ++s) {
    const int t = dir ? (T - 1 - s) : s;

    // load A tile (h) from LDS once: 8 K-steps of 32
    const _Float16* hp = &hst[mra][0] + hi * 8;
    v16h avs[8];
#pragma unroll
    for (int kk = 0; kk < 8; ++kk) {
      const v8h a0 = *(const v8h*)(hp + kk * 32);
      const v8h a1 = *(const v8h*)(hp + kk * 32 + 16);
      avs[kk] = SHUF16(a0, a1);
    }

    // z = h @ Whh^T + G[t] : 64 N-tiles, 16 waves, 4 tiles each
    for (int nt = wave; nt < 64; nt += 16) {
      const int ncol = nt * 16 + ml;
      const _Float16* __restrict__ Bp = Whh + (size_t)ncol * 256 + hi * 8;
      v8f acc = {};
#pragma unroll
      for (int kk = 0; kk < 8; ++kk) {
        const v8h b0 = *(const v8h*)(Bp + kk * 32);
        const v8h b1 = *(const v8h*)(Bp + kk * 32 + 16);
        const v16h bv = SHUF16(b0, b1);
        acc = __builtin_amdgcn_wmma_f32_16x16x32_f16(false, avs[kk], false, bv,
                                                     (short)0, acc, false, false);
      }
#pragma unroll
      for (int r = 0; r < 8; ++r) {
        const int m = r + 8 * hi;
        if (m < 4)
          zbuf[m][ncol] = acc[r] + G[((size_t)m * T + t) * 1024 + ncol];
      }
    }
    __syncthreads();

    for (int e = tid; e < 1024; e += 512) {
      const int b = e >> 8, j = e & 255;
      const float zi = zbuf[b][j];
      const float zf = zbuf[b][256 + j];
      const float zg = zbuf[b][512 + j];
      const float zo = zbuf[b][768 + j];
      const float cc = sigmf(zf) * cst[b][j] + sigmf(zi) * tanhf(zg);
      const float hh = sigmf(zo) * tanhf(cc);
      cst[b][j] = cc;
      hst[b][j] = (_Float16)hh;
      if (x1out)
        x1out[((size_t)b * T + t) * 512 + dir * 256 + j] = (_Float16)hh;
      if (s == T - 1)
        hs16[(size_t)b * 1024 + hsBase + dir * 256 + j] = (_Float16)hh;
    }
    __syncthreads();
  }
}

// ---------------------------------------------------------------------------
// Causal attention, one (b, head, q-row) per 64-thread block. S = 33.
// ---------------------------------------------------------------------------
__global__ void attn_kernel(const float* __restrict__ q,
                            const float* __restrict__ k,
                            const float* __restrict__ v,
                            _Float16* __restrict__ o16)
{
  __shared__ float sc[64];
  __shared__ float stat[2];

  const int idx = blockIdx.x;
  const int qi = idx % SEQ;
  const int h  = (idx / SEQ) % NHEADS;
  const int b  = idx / (SEQ * NHEADS);
  const int tk = threadIdx.x;

  if (tk < SEQ && tk <= qi) {
    const float* qp = q + ((size_t)(b * SEQ + qi)) * DTM + h * HDIM;
    const float* kp = k + ((size_t)(b * SEQ + tk)) * DTM + h * HDIM;
    float d = 0.f;
    for (int c = 0; c < HDIM; ++c) d += qp[c] * kp[c];
    sc[tk] = d * 0.0625f;                 // 1/sqrt(256)
  } else {
    sc[tk] = -1e30f;
  }
  __syncthreads();
  if (tk == 0) {
    float mx = -1e30f;
    for (int i = 0; i < 64; ++i) mx = fmaxf(mx, sc[i]);
    stat[0] = mx;
  }
  __syncthreads();
  const float p = expf(sc[tk] - stat[0]);
  sc[tk] = p;
  __syncthreads();
  if (tk == 0) {
    float sm = 0.f;
    for (int i = 0; i < 64; ++i) sm += sc[i];
    stat[1] = sm;
  }
  __syncthreads();
  const float inv = 1.f / stat[1];
#pragma unroll
  for (int dd = 0; dd < 4; ++dd) {
    const int d = tk + dd * 64;
    float acc = 0.f;
    for (int t2 = 0; t2 <= qi; ++t2)
      acc += sc[t2] * v[((size_t)(b * SEQ + t2)) * DTM + h * HDIM + d];
    o16[((size_t)(b * SEQ + qi)) * DTM + h * HDIM + d] = (_Float16)(acc * inv);
  }
}

// ---------------------------------------------------------------------------
// y = LN(x + t) with gain/bias; writes f32 + f16. One 512-wide row per block.
// ---------------------------------------------------------------------------
__global__ void add_ln(const float* __restrict__ x, const float* __restrict__ t,
                       const float* __restrict__ g, const float* __restrict__ be,
                       float* __restrict__ y32, _Float16* __restrict__ y16)
{
  __shared__ float red[128];
  const int row = blockIdx.x;
  const int tid = threadIdx.x;
  float s[4];
  float sum = 0.f, sq = 0.f;
#pragma unroll
  for (int u = 0; u < 4; ++u) {
    const int d = tid + u * 128;
    const float vv = x[(size_t)row * DTM + d] + t[(size_t)row * DTM + d];
    s[u] = vv; sum += vv; sq += vv * vv;
  }
  red[tid] = sum; __syncthreads();
  for (int st = 64; st > 0; st >>= 1) { if (tid < st) red[tid] += red[tid + st]; __syncthreads(); }
  const float mean = red[0] * (1.f / DTM);
  __syncthreads();
  red[tid] = sq; __syncthreads();
  for (int st = 64; st > 0; st >>= 1) { if (tid < st) red[tid] += red[tid + st]; __syncthreads(); }
  const float var = red[0] * (1.f / DTM) - mean * mean;
  const float inv = rsqrtf(var + 1e-5f);
#pragma unroll
  for (int u = 0; u < 4; ++u) {
    const int d = tid + u * 128;
    const float yy = (s[u] - mean) * inv * g[d] + be[d];
    y32[(size_t)row * DTM + d] = yy;
    y16[(size_t)row * DTM + d] = (_Float16)yy;
  }
}

// ---------------------------------------------------------------------------
// Small helper kernels
// ---------------------------------------------------------------------------
__global__ void zero_f16(_Float16* p, int n) {
  const int i = blockIdx.x * blockDim.x + threadIdx.x;
  if (i < n) p[i] = (_Float16)0;
}

// f32 -> f16 weight conversion with optional zero row padding
__global__ void cvt_w(const float* __restrict__ src, _Float16* __restrict__ dst,
                      int rows, int padRows, int cols)
{
  const int i = blockIdx.x * blockDim.x + threadIdx.x;
  if (i >= padRows * cols) return;
  const int r = i / cols;
  dst[i] = (r < rows) ? (_Float16)src[i] : (_Float16)0;
}

__global__ void embed_kernel(const int* __restrict__ mid,
                             const float* __restrict__ emb,
                             _Float16* __restrict__ x016)
{
  const int e = blockIdx.x * blockDim.x + threadIdx.x;   // B*T*256
  if (e >= BATCHN * TSEQ * DMODEL) return;
  const int r = e >> 8, c = e & 255;
  x016[e] = (_Float16)emb[(size_t)mid[r] * DMODEL + c];
}

__global__ void build_xin(const float* __restrict__ tmpT,
                          const float* __restrict__ latent,
                          float* __restrict__ xin32,
                          _Float16* __restrict__ xin16)
{
  const int e = blockIdx.x * blockDim.x + threadIdx.x;   // B*S*512
  if (e >= BATCHN * SEQ * DTM) return;
  const int row = e >> 9, c = e & 511;
  const int b = row / SEQ;
  const float v = (c < DMODEL) ? tmpT[(size_t)row * DMODEL + c]
                               : latent[(size_t)b * DMODEL + (c - DMODEL)];
  xin32[e] = v;
  xin16[e] = (_Float16)v;
}

__global__ void rowset(const float* __restrict__ y,
                       _Float16* __restrict__ buf16,
                       _Float16* __restrict__ dec16, int i)
{
  const int e = blockIdx.x * blockDim.x + threadIdx.x;   // B*512
  if (e >= BATCHN * DTM) return;
  const int b = e >> 9, c = e & 511;
  const float v = y[((size_t)b * SEQ + i) * DTM + c];
  buf16[((size_t)b * SEQ + i + 1) * DTM + c] = (_Float16)v;
  dec16[((size_t)b * MSTEPS + i) * DTM + c]  = (_Float16)v;
}

// ---------------------------------------------------------------------------
// Host-side orchestration
// ---------------------------------------------------------------------------
static inline void launch_gemm(const _Float16* A, const _Float16* Wh, const float* bias,
                               float* C32, _Float16* C16,
                               int M, int N, int K, int relu, hipStream_t stream)
{
  const int nTiles = (N + 15) / 16;
  dim3 grid((nTiles + 3) / 4, (M + 15) / 16);
  gemm_wmma<<<grid, 128, 0, stream>>>(A, Wh, bias, C32, C16, M, N, K, relu);
}

extern "C" void kernel_launch(void* const* d_in, const int* in_sizes, int n_in,
                              void* d_out, int out_size, void* d_ws, size_t ws_size,
                              hipStream_t stream)
{
  (void)in_sizes; (void)n_in; (void)out_size; (void)ws_size;
  auto F  = [&](int i) { return (const float*)d_in[i]; };
  const int* mid = (const int*)d_in[0];

  // workspace allocator
  char* wsb = (char*)d_ws;
  size_t off = 0;
  auto alloc = [&](size_t bytes) -> void* {
    void* p = wsb + off;
    off = (off + bytes + 255) & ~(size_t)255;
    return p;
  };
  auto allocW = [&](int padRows, int cols) -> _Float16* {
    return (_Float16*)alloc((size_t)padRows * cols * 2);
  };

  // f16 weight copies (rows padded to multiple of 16)
  _Float16* wih0f = allocW(1024, 256);
  _Float16* wih0b = allocW(1024, 256);
  _Float16* whh0f = allocW(1024, 256);
  _Float16* whh0b = allocW(1024, 256);
  _Float16* wih1f = allocW(1024, 512);
  _Float16* wih1b = allocW(1024, 512);
  _Float16* whh1f = allocW(1024, 256);
  _Float16* whh1b = allocW(1024, 256);
  _Float16* redw  = allocW(256, 1024);
  _Float16* dmodw = allocW(256, 512);
  _Float16* tw[2][4];          // q,k,v,o per layer
  _Float16* ff1w[2];
  _Float16* ff2w[2];
  for (int l = 0; l < 2; ++l) {
    for (int n = 0; n < 4; ++n) tw[l][n] = allocW(512, 512);
    ff1w[l] = allocW(2048, 512);
    ff2w[l] = allocW(512, 2048);
  }
  _Float16* fcw  = allocW(512, 512);
  _Float16* outw = allocW(400, 512);   // 388 padded to 400

  // activations
  float*    Gf    = (float*)alloc((size_t)1024 * 1024 * 4);
  float*    Gb    = (float*)alloc((size_t)1024 * 1024 * 4);
  _Float16* x016  = (_Float16*)alloc((size_t)BATCHN * TSEQ * DMODEL * 2);
  _Float16* x116  = (_Float16*)alloc((size_t)BATCHN * TSEQ * DTM * 2);
  _Float16* hs16  = (_Float16*)alloc((size_t)BATCHN * 1024 * 2);
  float*    latent= (float*)alloc((size_t)BATCHN * DMODEL * 4);
  _Float16* buf16 = (_Float16*)alloc((size_t)BATCHN * SEQ * DTM * 2);
  float*    tmpT  = (float*)alloc((size_t)BATCHN * SEQ * DMODEL * 4);
  float*    xA32  = (float*)alloc((size_t)BATCHN * SEQ * DTM * 4);
  _Float16* xA16  = (_Float16*)alloc((size_t)BATCHN * SEQ * DTM * 2);
  float*    xB32  = (float*)alloc((size_t)BATCHN * SEQ * DTM * 4);
  _Float16* xB16  = (_Float16*)alloc((size_t)BATCHN * SEQ * DTM * 2);
  float*    xC32  = (float*)alloc((size_t)BATCHN * SEQ * DTM * 4);
  _Float16* xC16  = (_Float16*)alloc((size_t)BATCHN * SEQ * DTM * 2);
  float*    qb    = (float*)alloc((size_t)BATCHN * SEQ * DTM * 4);
  float*    kbf   = (float*)alloc((size_t)BATCHN * SEQ * DTM * 4);
  float*    vbf   = (float*)alloc((size_t)BATCHN * SEQ * DTM * 4);
  _Float16* o16   = (_Float16*)alloc((size_t)BATCHN * SEQ * DTM * 2);
  float*    tpj   = (float*)alloc((size_t)BATCHN * SEQ * DTM * 4);
  _Float16* ffh16 = (_Float16*)alloc((size_t)BATCHN * SEQ * FFH * 2);
  float*    ybuf  = (float*)alloc((size_t)BATCHN * SEQ * DTM * 4);
  _Float16* dec16 = (_Float16*)alloc((size_t)BATCHN * MSTEPS * DTM * 2);

  auto cvt = [&](const float* src, _Float16* dst, int rows, int padRows, int cols) {
    const int total = padRows * cols;
    cvt_w<<<(total + 255) / 256, 256, 0, stream>>>(src, dst, rows, padRows, cols);
  };

  // --- weight conversion (f32 -> f16, padded) ----------------------------
  cvt(F(2),  wih0f, 1024, 1024, 256);
  cvt(F(3),  whh0f, 1024, 1024, 256);
  cvt(F(5),  wih0b, 1024, 1024, 256);
  cvt(F(6),  whh0b, 1024, 1024, 256);
  cvt(F(8),  wih1f, 1024, 1024, 512);
  cvt(F(9),  whh1f, 1024, 1024, 256);
  cvt(F(11), wih1b, 1024, 1024, 512);
  cvt(F(12), whh1b, 1024, 1024, 256);
  cvt(F(14), redw,  256, 256, 1024);
  cvt(F(16), dmodw, 256, 256, 512);
  for (int l = 0; l < 2; ++l) {
    const int bs = 18 + l * 16;
    for (int n = 0; n < 4; ++n) cvt(F(bs + 2 * n), tw[l][n], 512, 512, 512);
    cvt(F(bs + 12), ff1w[l], 2048, 2048, 512);
    cvt(F(bs + 14), ff2w[l], 512, 512, 2048);
  }
  cvt(F(50), fcw,  512, 512, 512);
  cvt(F(52), outw, 388, 400, 512);

  const int MROWS = BATCHN * SEQ;   // 132

  // --- init + embedding --------------------------------------------------
  {
    const int n = BATCHN * SEQ * DTM;
    zero_f16<<<(n + 255) / 256, 256, 0, stream>>>(buf16, n);
    const int ne = BATCHN * TSEQ * DMODEL;
    embed_kernel<<<(ne + 255) / 256, 256, 0, stream>>>(mid, F(1), x016);
  }

  // --- LSTM layer 0 ------------------------------------------------------
  launch_gemm(x016, wih0f, F(4), Gf, nullptr, 1024, 1024, 256, 0, stream);
  launch_gemm(x016, wih0b, F(7), Gb, nullptr, 1024, 1024, 256, 0, stream);
  lstm_rec<<<2, 512, 0, stream>>>(Gf, Gb, whh0f, whh0b, x116, hs16, 0, TSEQ);

  // --- LSTM layer 1 ------------------------------------------------------
  launch_gemm(x116, wih1f, F(10), Gf, nullptr, 1024, 1024, 512, 0, stream);
  launch_gemm(x116, wih1b, F(13), Gb, nullptr, 1024, 1024, 512, 0, stream);
  lstm_rec<<<2, 512, 0, stream>>>(Gf, Gb, whh1f, whh1b, nullptr, hs16, 512, TSEQ);

  // --- latent ------------------------------------------------------------
  launch_gemm(hs16, redw, F(15), latent, nullptr, BATCHN, DMODEL, 1024, 0, stream);

  // --- autoregressive decode (no KV cache, full re-run each step) --------
  for (int i = 0; i < MSTEPS; ++i) {
    launch_gemm(buf16, dmodw, F(17), tmpT, nullptr, MROWS, DMODEL, DTM, 0, stream);
    {
      const int n = BATCHN * SEQ * DTM;
      build_xin<<<(n + 255) / 256, 256, 0, stream>>>(tmpT, latent, xA32, xA16);
    }
    const float*    in32 = xA32;
    const _Float16* in16 = xA16;
    for (int l = 0; l < 2; ++l) {
      const int bs = 18 + l * 16;
      launch_gemm(in16, tw[l][0], F(bs + 1), qb,  nullptr, MROWS, DTM, DTM, 0, stream);
      launch_gemm(in16, tw[l][1], F(bs + 3), kbf, nullptr, MROWS, DTM, DTM, 0, stream);
      launch_gemm(in16, tw[l][2], F(bs + 5), vbf, nullptr, MROWS, DTM, DTM, 0, stream);
      attn_kernel<<<BATCHN * NHEADS * SEQ, 64, 0, stream>>>(qb, kbf, vbf, o16);
      launch_gemm(o16, tw[l][3], F(bs + 7), tpj, nullptr, MROWS, DTM, DTM, 0, stream);
      add_ln<<<MROWS, 128, 0, stream>>>(in32, tpj, F(bs + 8), F(bs + 9), xC32, xC16);
      launch_gemm(xC16, ff1w[l], F(bs + 13), nullptr, ffh16, MROWS, FFH, DTM, 1, stream);
      launch_gemm(ffh16, ff2w[l], F(bs + 15), tpj, nullptr, MROWS, DTM, FFH, 0, stream);
      float*    out32 = (l == 0) ? xB32 : xA32;
      _Float16* out16 = (l == 0) ? xB16 : xA16;
      add_ln<<<MROWS, 128, 0, stream>>>(xC32, tpj, F(bs + 10), F(bs + 11), out32, out16);
      in32 = out32; in16 = out16;
    }
    launch_gemm(in16, fcw, F(51), ybuf, nullptr, MROWS, DTM, DTM, 0, stream);
    {
      const int n = BATCHN * DTM;
      rowset<<<(n + 255) / 256, 256, 0, stream>>>(ybuf, buf16, dec16, i);
    }
  }

  // --- output projection: (B*32, 512) x (388, 512)^T + out_b -------------
  launch_gemm(dec16, outw, F(53), (float*)d_out, nullptr,
              BATCHN * MSTEPS, NVOCAB, DTM, 0, stream);
}
